// GCNEncoder_17669495456113
// MI455X (gfx1250) — compile-verified, matrix operand order checked by reading, not run on hardware
//
#include <hip/hip_runtime.h>
#include <hip/hip_bf16.h>

typedef __attribute__((ext_vector_type(16))) __bf16 v16bf;
typedef __attribute__((ext_vector_type(8)))  float  v8f;

#define MT 4  // M-tiles per wave (B-fragment reuse factor)

// ---------------------------------------------------------------- utilities

__global__ __launch_bounds__(256) void k_zero_f32(float* __restrict__ p, long n) {
    long i = (long)blockIdx.x * blockDim.x + threadIdx.x;
    long stride = (long)gridDim.x * blockDim.x;
    for (; i < n; i += stride) p[i] = 0.0f;
}

__global__ __launch_bounds__(256) void k_cvt_f32_bf16(const float* __restrict__ in,
                                                      __bf16* __restrict__ out, long n) {
    long i = (long)blockIdx.x * blockDim.x + threadIdx.x;
    long stride = (long)gridDim.x * blockDim.x;
    for (; i < n; i += stride) out[i] = (__bf16)in[i];
}

// deg[dst] += 1 for every edge (self-loop added later as +1)
__global__ __launch_bounds__(256) void k_degree(const int* __restrict__ dst,
                                                float* __restrict__ deg, int E) {
    int e = blockIdx.x * blockDim.x + threadIdx.x;
    if (e < E) atomicAdd(&deg[dst[e]], 1.0f);
}

// deg -> rsqrt(deg + 1)  (in place)
__global__ __launch_bounds__(256) void k_dinv(float* __restrict__ deg, int n) {
    int i = blockIdx.x * blockDim.x + threadIdx.x;
    if (i < n) deg[i] = rsqrtf(deg[i] + 1.0f);
}

// ------------------------------------------- weight pre-swizzle (f32 -> bf16)
// Packs W[K,N] (row major f32) into WMMA-B fragment-major bf16:
//   Bp[(((t*kchunks + q)*32 + lane)*16 + j]  = element j of lane's v16bf for
//   n-tile t, k-chunk q.  Fragment order: j -> (vgpr j>>1, half j&1),
//   kb = ((r&4)<<2) | (lanehalf<<3) | ((r&3)<<1) | h   (ISA 16-bit B layout).
__global__ __launch_bounds__(256) void k_pack_w(const float* __restrict__ W,
                                                __bf16* __restrict__ Bp,
                                                int N, int K) {
    const int kchunks = K >> 5;
    long idx = (long)blockIdx.x * blockDim.x + threadIdx.x;
    long total = (long)N * K;
    if (idx >= total) return;
    const int j = (int)(idx & 15);
    const int l = (int)((idx >> 4) & 31);
    long rest   = idx >> 9;
    const int q = (int)(rest % kchunks);
    const int t = (int)(rest / kchunks);
    const int r = j >> 1, h = j & 1;
    const int kb = ((r & 4) << 2) + ((l >> 4) << 3) + ((r & 3) << 1) + h;
    const int k  = q * 32 + kb;
    const int n  = t * 16 + (l & 15);
    Bp[idx] = (__bf16)W[(size_t)k * N + n];
}

// ------------------------------------------------------- WMMA bf16 GEMM
// C[M,N] (f32) = A[M,K] (bf16 row major) * W (pre-swizzled bf16 fragments)
// Each wave: MT consecutive 16-row tiles x one 16-col tile, sharing B frags.
// grid = (ceil(M/(16*MT)), N/128), block = 256 (8 waves).  N % 128 == 0.
__global__ __launch_bounds__(256) void k_gemm_bf16(const __bf16* __restrict__ A,
                                                   const __bf16* __restrict__ Bp,
                                                   float* __restrict__ C,
                                                   int M, int N, int K) {
    const int lane  = threadIdx.x & 31;
    const int wave  = threadIdx.x >> 5;
    const int ntile = blockIdx.y * 8 + wave;
    const int mbase = blockIdx.x * (16 * MT);

    const int m     = lane & 15;
    const int khalf = lane >> 4;
    const int ncol  = ntile * 16 + (lane & 15);
    const int kchunks = K >> 5;

    const __bf16* __restrict__ Arow[MT];
#pragma unroll
    for (int t = 0; t < MT; ++t) {
        int r = mbase + t * 16 + m;
        if (r >= M) r = M - 1;                 // clamped loads; stores guarded below
        Arow[t] = A + (size_t)r * K;
    }
    const __bf16* __restrict__ Bbase =
        Bp + ((size_t)ntile * kchunks * 32 + lane) * 16;

    v8f acc[MT] = {};
    for (int q = 0; q < kchunks; ++q) {
        const v16bf b = *(const v16bf*)(Bbase + (size_t)q * 512);  // 32B aligned
        const int koff = q * 32 + (khalf << 3);
#pragma unroll
        for (int t = 0; t < MT; ++t) {
            v16bf a;
#pragma unroll
            for (int j = 0; j < 8; ++j) {      // two contiguous 16B groups -> 2x b128
                a[j]     = Arow[t][koff + j];
                a[8 + j] = Arow[t][koff + 16 + j];
            }
            acc[t] = __builtin_amdgcn_wmma_f32_16x16x32_bf16(
                false, a, false, b, (short)0, acc[t], false, false);
        }
    }

    if (mbase + 16 * MT <= M) {                // wave-uniform fast path: no guards
#pragma unroll
        for (int t = 0; t < MT; ++t)
#pragma unroll
            for (int r = 0; r < 8; ++r)
                C[(size_t)(mbase + t * 16 + r + (khalf << 3)) * N + ncol] = acc[t][r];
    } else {
#pragma unroll
        for (int t = 0; t < MT; ++t)
#pragma unroll
            for (int r = 0; r < 8; ++r) {
                const int row = mbase + t * 16 + r + (khalf << 3);
                if (row < M) C[(size_t)row * N + ncol] = acc[t][r];
            }
    }
}

// ---------------------------------------------- edge scatter-add aggregation
// one wave per edge: agg[dst][c] += h[src][c] * dinv[src]*dinv[dst]
// h and agg are L2-resident (51 MB each << 192 MB L2) -> atomics resolve in L2.
__global__ __launch_bounds__(256) void k_edge_agg(const int* __restrict__ src,
                                                  const int* __restrict__ dst,
                                                  const float* __restrict__ dinv,
                                                  const float* __restrict__ h,
                                                  float* __restrict__ agg,
                                                  int E, int C) {
    const int wid  = (int)((blockIdx.x * (unsigned)blockDim.x + threadIdx.x) >> 5);
    const int lane = threadIdx.x & 31;
    if (wid >= E) return;
    const int s = src[wid];
    const int d = dst[wid];
    const float norm = dinv[s] * dinv[d];
    const float* __restrict__ hrow = h + (size_t)s * C;
    float* __restrict__ arow       = agg + (size_t)d * C;
    for (int c = lane; c < C; c += 32)
        atomicAdd(&arow[c], hrow[c] * norm);
}

// hidden = relu(agg + h*dinv^2 + b)  ->  bf16 for the next WMMA GEMM
__global__ __launch_bounds__(256) void k_finalize_relu_bf16(const float* __restrict__ agg,
                                                            const float* __restrict__ h,
                                                            const float* __restrict__ dinv,
                                                            const float* __restrict__ bias,
                                                            __bf16* __restrict__ out,
                                                            int Nn, int C) {
    long idx = (long)blockIdx.x * blockDim.x + threadIdx.x;
    long total = (long)Nn * C;
    if (idx >= total) return;
    const int i = (int)(idx / C);
    const int c = (int)(idx % C);
    const float di = dinv[i];
    float v = agg[idx] + h[idx] * di * di + bias[c];
    v = fmaxf(v, 0.0f);
    out[idx] = (__bf16)v;
}

// out += h*dinv^2 + b   (out already holds the scattered neighbor sum)
__global__ __launch_bounds__(256) void k_finalize_out(float* __restrict__ out,
                                                      const float* __restrict__ h,
                                                      const float* __restrict__ dinv,
                                                      const float* __restrict__ bias,
                                                      int Nn, int C) {
    long idx = (long)blockIdx.x * blockDim.x + threadIdx.x;
    long total = (long)Nn * C;
    if (idx >= total) return;
    const int i = (int)(idx / C);
    const int c = (int)(idx % C);
    const float di = dinv[i];
    out[idx] += h[idx] * di * di + bias[c];
}

// ---------------------------------------------------------------- launcher

extern "C" void kernel_launch(void* const* d_in, const int* in_sizes, int n_in,
                              void* d_out, int out_size, void* d_ws, size_t ws_size,
                              hipStream_t stream) {
    const float* x  = (const float*)d_in[0];
    const int*   ei = (const int*)  d_in[1];
    const float* W1 = (const float*)d_in[2];
    const float* b1 = (const float*)d_in[3];
    const float* W2 = (const float*)d_in[4];
    const float* b2 = (const float*)d_in[5];
    float* out = (float*)d_out;

    // Derive shapes (reference: x[N,IN], ei[2,E], W1[IN,HID], b1[HID], W2[HID,OUT], b2[OUT])
    const int HID = in_sizes[3];
    const int OUT = in_sizes[5];
    const int IN  = in_sizes[2] / HID;
    const int Nn  = in_sizes[0] / IN;
    const int E   = in_sizes[1] / 2;

    const int* src = ei;
    const int* dst = ei + E;

    // bump-allocate scratch
    size_t off = 0;
    auto alloc = [&](size_t bytes) -> void* {
        void* p = (char*)d_ws + off;
        off += (bytes + 255) & ~(size_t)255;
        return p;
    };
    float*  dinv  = (float*) alloc((size_t)Nn * 4);            // deg, then rsqrt(deg+1)
    float*  h1    = (float*) alloc((size_t)Nn * HID * 4);      // x @ W1
    float*  agg1  = (float*) alloc((size_t)Nn * HID * 4);      // neighbor sum, layer 1
    float*  h2    = (float*) alloc((size_t)Nn * OUT * 4);      // hidden @ W2
    __bf16* xbf   = (__bf16*)alloc((size_t)Nn * IN * 2);
    __bf16* hidbf = (__bf16*)alloc((size_t)Nn * HID * 2);
    __bf16* w1p   = (__bf16*)alloc((size_t)IN * HID * 2);      // pre-swizzled fragments
    __bf16* w2p   = (__bf16*)alloc((size_t)HID * OUT * 2);
    (void)ws_size; (void)n_in; (void)out_size;

    const int TPB = 256;
    auto cdiv = [](long a, long b) { return (int)((a + b - 1) / b); };

    // 1. degrees -> dinv
    k_zero_f32<<<cdiv(Nn, TPB), TPB, 0, stream>>>(dinv, Nn);
    k_degree  <<<cdiv(E, TPB), TPB, 0, stream>>>(dst, dinv, E);
    k_dinv    <<<cdiv(Nn, TPB), TPB, 0, stream>>>(dinv, Nn);

    // 2. bf16 x; fragment-packed bf16 weights
    k_cvt_f32_bf16<<<cdiv((long)Nn * IN, TPB), TPB, 0, stream>>>(x, xbf, (long)Nn * IN);
    k_pack_w<<<cdiv((long)IN * HID, TPB), TPB, 0, stream>>>(W1, w1p, HID, IN);
    k_pack_w<<<cdiv((long)HID * OUT, TPB), TPB, 0, stream>>>(W2, w2p, OUT, HID);

    // 3. h1 = x @ W1   (WMMA bf16)
    {
        dim3 grid(cdiv(Nn, 16 * MT), HID / 128);
        k_gemm_bf16<<<grid, TPB, 0, stream>>>(xbf, w1p, h1, Nn, HID, IN);
    }

    // 4. layer-1 neighbor aggregation
    k_zero_f32<<<cdiv((long)Nn * HID, TPB), TPB, 0, stream>>>(agg1, (long)Nn * HID);
    k_edge_agg<<<cdiv((long)E * 32, TPB), TPB, 0, stream>>>(src, dst, dinv, h1, agg1, E, HID);

    // 5. hidden = relu(agg1 + h1*dinv^2 + b1) -> bf16
    k_finalize_relu_bf16<<<cdiv((long)Nn * HID, TPB), TPB, 0, stream>>>(
        agg1, h1, dinv, b1, hidbf, Nn, HID);

    // 6. h2 = hidden @ W2  (WMMA bf16)
    {
        dim3 grid(cdiv(Nn, 16 * MT), OUT / 128);
        k_gemm_bf16<<<grid, TPB, 0, stream>>>(hidbf, w2p, h2, Nn, OUT, HID);
    }

    // 7. layer-2 neighbor aggregation straight into d_out
    k_zero_f32<<<cdiv((long)Nn * OUT, TPB), TPB, 0, stream>>>(out, (long)Nn * OUT);
    k_edge_agg<<<cdiv((long)E * 32, TPB), TPB, 0, stream>>>(src, dst, dinv, h2, out, E, OUT);

    // 8. out += h2*dinv^2 + b2
    k_finalize_out<<<cdiv((long)Nn * OUT, TPB), TPB, 0, stream>>>(out, h2, dinv, b2, Nn, OUT);
}
